// SINGLEModel_49228915147211
// MI455X (gfx1250) — compile-verified, matrix operand order checked by reading, not run on hardware
//
#include <hip/hip_runtime.h>
#include <hip/hip_bf16.h>

typedef __attribute__((ext_vector_type(8)))  _Float16 v8h;
typedef __attribute__((ext_vector_type(16))) _Float16 v16h;
typedef __attribute__((ext_vector_type(8)))  float    v8f;
typedef unsigned int u32x4 __attribute__((ext_vector_type(4)));
typedef int          i32x4 __attribute__((ext_vector_type(4)));
typedef int          i32x8 __attribute__((ext_vector_type(8)));

#define POMO  100
#define NN    101
#define DD    128
#define NH    8
#define DKH   16
#define MT    7        // 112/16 row tiles (POMO and N padded to 112)
#define PAD_R 112
#define SSTR  116      // f32 score-row stride

// f16 weight layout inside d_ws (element = _Float16)
#define WS_K  0
#define WS_V  16384
#define WS_Q  32768
#define WS_C  49152
#define WS_TOTAL 65536

// LDS byte offsets of the two weight buffers (see layout in kernel)
#define LDS_WA_OFF 176128u   // (5*14336 + 16384) halves * 2
#define LDS_WB_OFF 208896u   // LDS_WA_OFF + 32768

#if defined(__HIP_DEVICE_COMPILE__) && __has_builtin(__builtin_amdgcn_tensor_load_to_lds)
#define HAVE_TDM 1
#else
#define HAVE_TDM 0
#endif

// ---------------------------------------------------------------------------
// WMMA helpers (CDNA5 wave32, V_WMMA_F32_16X16X32_F16)
// ---------------------------------------------------------------------------
__device__ __forceinline__ v8f wmma_f16(v16h a, v16h b, v8f c) {
  return __builtin_amdgcn_wmma_f32_16x16x32_f16(false, a, false, b, (short)0, c,
                                                false, false);
}

// A/B fragment of a 16x32 (or 32x16) f16 tile from row-major LDS.
__device__ __forceinline__ v16h load_frag(const _Float16* base, int stride,
                                          int row_base, int k_base, int lane) {
  const int r  = row_base + (lane & 15);
  const int ko = (lane >> 4) << 3;
  const _Float16* p = base + r * stride + k_base + ko;
  v8h lo = *(const v8h*)(p);
  v8h hi = *(const v8h*)(p + 16);
  return __builtin_shufflevector(lo, hi, 0,1,2,3,4,5,6,7,8,9,10,11,12,13,14,15);
}

// Effective K = 16 (upper K half zeroed) for the DK=16 attention-score GEMM.
__device__ __forceinline__ v16h load_frag_k16(const _Float16* base, int stride,
                                              int row_base, int k_base, int lane) {
  const int r  = row_base + (lane & 15);
  const int ko = (lane >> 4) << 3;
  const _Float16* p = base + r * stride + k_base + ko;
  v8h lo = *(const v8h*)(p);
  v8h hi = {};
  return __builtin_shufflevector(lo, hi, 0,1,2,3,4,5,6,7,8,9,10,11,12,13,14,15);
}

__device__ __forceinline__ v8f gemm_acc_128(const _Float16* A, const _Float16* B,
                                            int mi, int ni, int lane) {
  v8f acc = {};
#pragma unroll
  for (int kt = 0; kt < 4; ++kt) {
    v16h a = load_frag(A, DD, mi * 16, kt * 32, lane);
    v16h b = load_frag(B, DD, ni * 16, kt * 32, lane);
    acc = wmma_f16(a, b, acc);
  }
  return acc;
}

__device__ __forceinline__ float wave_max(float v) {
#pragma unroll
  for (int o = 16; o > 0; o >>= 1) v = fmaxf(v, __shfl_xor(v, o, 32));
  return v;
}
__device__ __forceinline__ float wave_sum(float v) {
#pragma unroll
  for (int o = 16; o > 0; o >>= 1) v += __shfl_xor(v, o, 32);
  return v;
}

// ---------------------------------------------------------------------------
// TDM: DMA one contiguous 32 KB f16 weight matrix from global into LDS.
// D# per CDNA5 ISA ch.8: data_size=8B, tensor/tile dim0 = 4096 units, 1 row.
// ---------------------------------------------------------------------------
#if HAVE_TDM
__device__ __forceinline__ void tdm_copy_32k(unsigned lds_byte_off,
                                             const _Float16* gsrc) {
  const unsigned long long ga = (unsigned long long)(size_t)gsrc;
  const unsigned lds_addr = __builtin_amdgcn_groupstaticsize() + lds_byte_off;
  u32x4 g0 = {
      1u,                                                 // count=1 (valid, user)
      lds_addr,                                           // lds_addr
      (unsigned)(ga & 0xFFFFFFFFull),                     // global_addr[31:0]
      (unsigned)((ga >> 32) & 0x1FFFFFFull) | (2u << 30)  // ga[56:32] | type=2
  };
  i32x8 g1 = {
      (int)(3u << 16),           // workgroup_mask=0, data_size=3 (8B)
      (int)(4096u << 16),        // tensor_dim0[15:0] << 16 (abar addr = 0)
      (int)(1u << 16),           // tensor_dim0[31:16]=0 | tensor_dim1[15:0]=1
      (int)(4096u << 16),        // tensor_dim1 hi=0 | tile_dim0=4096
      1,                         // tile_dim1=1, tile_dim2=0
      4096,                      // tensor_dim0_stride[31:0]
      0, 0                       // stride hi / tensor_dim1_stride
  };
  i32x4 z4 = {0, 0, 0, 0};
#if __clang_major__ >= 23
  i32x8 z8 = {0, 0, 0, 0, 0, 0, 0, 0};
  __builtin_amdgcn_tensor_load_to_lds(g0, g1, z4, z4, z8, 0);
#else
  __builtin_amdgcn_tensor_load_to_lds(g0, g1, z4, z4, 0);
#endif
}
#endif

// ---------------------------------------------------------------------------
// Prep: convert the four weight matrices to f16 in d_ws (shared by all blocks).
// ---------------------------------------------------------------------------
__global__ void convert_weights_f16(const float* __restrict__ Wq,
                                    const float* __restrict__ Wk,
                                    const float* __restrict__ Wv,
                                    const float* __restrict__ Wc,
                                    _Float16* __restrict__ ws) {
  int i = blockIdx.x * blockDim.x + threadIdx.x;
  if (i < DD * DD) {
    int r = i >> 7, c = i & 127;
    ws[WS_K + i] = (_Float16)Wk[i];
    ws[WS_V + i] = (_Float16)Wv[i];
    ws[WS_C + i] = (_Float16)Wc[i];
    ws[WS_Q + i] = (_Float16)Wq[r * (DD + 1) + c];  // first 128 cols of Wq_last
  }
}

// ---------------------------------------------------------------------------
// Main: one workgroup (256 threads = 8 wave32) per batch element.
// ---------------------------------------------------------------------------
__global__ __launch_bounds__(256, 1)
void pomo_decoder_wmma(const float* __restrict__ enc,    // [B,N,D]
                       const float* __restrict__ lastn,  // [B,POMO,D]
                       const float* __restrict__ attr,   // [B,POMO]
                       const float* __restrict__ mask,   // [B,POMO,N]
                       const float* __restrict__ Wq,     // [D,D+1] (f32, attr col)
                       const float* __restrict__ bc,     // [D]
                       const _Float16* __restrict__ wsW, // f16 weights in d_ws
                       float* __restrict__ out)          // [B,POMO,N]
{
  extern __shared__ _Float16 smem[];
  _Float16* sE  = smem;                 // 112x128  encoded nodes (f16)
  _Float16* sK  = sE  + PAD_R * DD;     // 112x128  K
  _Float16* sVt = sK  + PAD_R * DD;     // 128x128  V transposed [d][n]
  _Float16* sQ  = sVt + DD * DD;        // 112x128  Q, later mh
  _Float16* sO  = sQ  + PAD_R * DD;     // 112x128  attention out (concat heads)
  _Float16* sP  = sO  + PAD_R * DD;     // 112x128  last-node input, then probs
  _Float16* sWa = sP  + PAD_R * DD;     // 128x128  weight buffer A (byte off 176128)
  _Float16* sWb = sWa + DD * DD;        // 128x128  weight buffer B (byte off 208896)
  float*    sS  = (float*)(sWb + DD * DD); // 112x116 f32 scores

  const int b    = blockIdx.x;
  const int tid  = threadIdx.x;
  const int wave = tid >> 5;
  const int lane = tid & 31;

  const float* Eb = enc  + (size_t)b * NN * DD;
  const float* Mb = mask + (size_t)b * POMO * NN;

  // ---- phase 0: TDM Wk -> A overlaps activation staging -------------------
#if HAVE_TDM
  if (wave == 0) tdm_copy_32k(LDS_WA_OFF, wsW + WS_K);
#endif
  for (int i = tid; i < PAD_R * DD; i += 256) {
    int r = i >> 7, c = i & 127;
    sE[i] = (_Float16)((r < NN) ? Eb[r * DD + c] : 0.f);
  }
  {
    const float* Lb = lastn + (size_t)b * POMO * DD;
    for (int i = tid; i < PAD_R * DD; i += 256) {
      int r = i >> 7, c = i & 127;
      sP[i] = (_Float16)((r < POMO) ? Lb[r * DD + c] : 0.f);
    }
  }
  for (int i = tid; i < DD * DD; i += 256) sVt[i] = (_Float16)0.f;
#if HAVE_TDM
  __builtin_amdgcn_s_wait_tensorcnt(0);
#else
  for (int i = tid; i < DD * DD; i += 256) sWa[i] = wsW[WS_K + i];
#endif
  __syncthreads();
#if HAVE_TDM
  if (wave == 0) tdm_copy_32k(LDS_WB_OFF, wsW + WS_V);  // overlaps K GEMM
#endif

  // ---- K = E @ Wk^T -------------------------------------------------------
  for (int t = wave; t < MT * 8; t += 8) {
    int mi = t >> 3, ni = t & 7;
    v8f acc = gemm_acc_128(sE, sWa, mi, ni, lane);
    int col = ni * 16 + (lane & 15);
    int rb  = mi * 16 + ((lane & 16) ? 8 : 0);
#pragma unroll
    for (int i = 0; i < 8; ++i) sK[(rb + i) * DD + col] = (_Float16)acc[i];
  }
  __syncthreads();
#if HAVE_TDM
  __builtin_amdgcn_s_wait_tensorcnt(0);   // Wv arrived in B
  __syncthreads();                        // A free for reuse
  if (wave == 0) tdm_copy_32k(LDS_WA_OFF, wsW + WS_Q);  // overlaps V GEMM
#else
  for (int i = tid; i < DD * DD; i += 256) sWb[i] = wsW[WS_V + i];
  __syncthreads();
#endif

  // ---- V = E @ Wv^T (stored transposed) -----------------------------------
  for (int t = wave; t < MT * 8; t += 8) {
    int mi = t >> 3, ni = t & 7;
    v8f acc = gemm_acc_128(sE, sWb, mi, ni, lane);
    int col = ni * 16 + (lane & 15);
    int rb  = mi * 16 + ((lane & 16) ? 8 : 0);
#pragma unroll
    for (int i = 0; i < 8; ++i) sVt[col * DD + (rb + i)] = (_Float16)acc[i];
  }
  __syncthreads();
#if HAVE_TDM
  __builtin_amdgcn_s_wait_tensorcnt(0);   // Wq arrived in A
  __syncthreads();                        // B free for reuse
  if (wave == 0) tdm_copy_32k(LDS_WB_OFF, wsW + WS_C);  // overlaps Q + attention
#else
  for (int i = tid; i < DD * DD; i += 256) sWa[i] = wsW[WS_Q + i];
  __syncthreads();
#endif

  // ---- Q = last @ Wq^T ----------------------------------------------------
  for (int t = wave; t < MT * 8; t += 8) {
    int mi = t >> 3, ni = t & 7;
    v8f acc = gemm_acc_128(sP, sWa, mi, ni, lane);
    int col = ni * 16 + (lane & 15);
    int rb  = mi * 16 + ((lane & 16) ? 8 : 0);
#pragma unroll
    for (int i = 0; i < 8; ++i) sQ[(rb + i) * DD + col] = (_Float16)acc[i];
  }
  __syncthreads();

  // ---- rank-1 attr term: q[p,d] += attr[p] * Wq[d,128] --------------------
  {
    const float* Ab = attr + (size_t)b * POMO;
    for (int i = tid; i < PAD_R * DD; i += 256) {
      int r = i >> 7, c = i & 127;
      if (r < POMO)
        sQ[i] = (_Float16)((float)sQ[i] + Ab[r] * Wq[c * (DD + 1) + DD]);
    }
  }
  __syncthreads();

  // ---- 8-head attention (DK=16) -------------------------------------------
  for (int h = 0; h < NH; ++h) {
    // score = (Q_h @ K_h^T) / sqrt(16)
    for (int t = wave; t < MT * MT; t += 8) {
      int mi = t / MT, ni = t % MT;
      v16h a  = load_frag_k16(sQ, DD, mi * 16, h * DKH, lane);
      v16h bb = load_frag_k16(sK, DD, ni * 16, h * DKH, lane);
      v8f acc = {};
      acc = wmma_f16(a, bb, acc);
      int col = ni * 16 + (lane & 15);
      int rb  = mi * 16 + ((lane & 16) ? 8 : 0);
#pragma unroll
      for (int i = 0; i < 8; ++i) sS[(rb + i) * SSTR + col] = acc[i] * 0.25f;
    }
    __syncthreads();

    // masked softmax over N -> probs f16 (zero-padded cols)
    for (int p = wave; p < PAD_R; p += 8) {
      float vals[4];
      float mx = -3.0e38f;
#pragma unroll
      for (int i = 0; i < 4; ++i) {
        int c = lane + i * 32;
        float v = -3.0e38f;
        if (c < NN) {
          v = sS[p * SSTR + c];
          if (p < POMO) v += Mb[p * NN + c];
        }
        vals[i] = v;
        mx = fmaxf(mx, v);
      }
      mx = wave_max(mx);
      float sum = 0.f;
#pragma unroll
      for (int i = 0; i < 4; ++i) {
        float e = (vals[i] > -1.0e38f) ? __expf(vals[i] - mx) : 0.f;
        vals[i] = e;
        sum += e;
      }
      sum = wave_sum(sum);
      float inv = 1.0f / sum;
#pragma unroll
      for (int i = 0; i < 4; ++i) {
        int c = lane + i * 32;
        sP[p * DD + c] = (_Float16)((c < NN) ? vals[i] * inv : 0.f);
      }
    }
    __syncthreads();

    // out_h = P @ V_h  (Vt rows contiguous in node index)
    for (int t = wave; t < MT; t += 8) {
      v8f acc = {};
#pragma unroll
      for (int kt = 0; kt < 4; ++kt) {
        v16h a  = load_frag(sP, DD, t * 16, kt * 32, lane);
        v16h bb = load_frag(sVt, DD, h * DKH, kt * 32, lane);
        acc = wmma_f16(a, bb, acc);
      }
      int col = h * DKH + (lane & 15);
      int rb  = t * 16 + ((lane & 16) ? 8 : 0);
#pragma unroll
      for (int i = 0; i < 8; ++i) sO[(rb + i) * DD + col] = (_Float16)acc[i];
    }
    __syncthreads();
  }

  // ---- mh = O @ Wc^T + b  (into sQ) ---------------------------------------
#if HAVE_TDM
  __builtin_amdgcn_s_wait_tensorcnt(0);   // Wc arrived in B (long ago)
  __syncthreads();
#else
  for (int i = tid; i < DD * DD; i += 256) sWb[i] = wsW[WS_C + i];
  __syncthreads();
#endif
  for (int t = wave; t < MT * 8; t += 8) {
    int mi = t >> 3, ni = t & 7;
    v8f acc = gemm_acc_128(sO, sWb, mi, ni, lane);
    int col = ni * 16 + (lane & 15);
    int rb  = mi * 16 + ((lane & 16) ? 8 : 0);
    float bias = bc[col];
#pragma unroll
    for (int i = 0; i < 8; ++i) sQ[(rb + i) * DD + col] = (_Float16)(acc[i] + bias);
  }
  __syncthreads();

  // ---- score2 = mh @ E^T / sqrt(128) --------------------------------------
  for (int t = wave; t < MT * MT; t += 8) {
    int mi = t / MT, ni = t % MT;
    v8f acc = {};
#pragma unroll
    for (int kt = 0; kt < 4; ++kt) {
      v16h a  = load_frag(sQ, DD, mi * 16, kt * 32, lane);
      v16h bb = load_frag(sE, DD, ni * 16, kt * 32, lane);
      acc = wmma_f16(a, bb, acc);
    }
    int col = ni * 16 + (lane & 15);
    int rb  = mi * 16 + ((lane & 16) ? 8 : 0);
#pragma unroll
    for (int i = 0; i < 8; ++i)
      sS[(rb + i) * SSTR + col] = acc[i] * 0.08838834764831845f;  // 1/sqrt(128)
  }
  __syncthreads();

  // ---- logits = 10*tanh(score2)+mask; softmax over N; write probs ---------
  float* Ob = out + (size_t)b * POMO * NN;
  for (int p = wave; p < POMO; p += 8) {
    float vals[4];
    float mx = -3.0e38f;
#pragma unroll
    for (int i = 0; i < 4; ++i) {
      int c = lane + i * 32;
      float v = -3.0e38f;
      if (c < NN) v = 10.0f * tanhf(sS[p * SSTR + c]) + Mb[p * NN + c];
      vals[i] = v;
      mx = fmaxf(mx, v);
    }
    mx = wave_max(mx);
    float sum = 0.f;
#pragma unroll
    for (int i = 0; i < 4; ++i) {
      float e = (vals[i] > -1.0e38f) ? __expf(vals[i] - mx) : 0.f;
      vals[i] = e;
      sum += e;
    }
    sum = wave_sum(sum);
    float inv = 1.0f / sum;
#pragma unroll
    for (int i = 0; i < 4; ++i) {
      int c = lane + i * 32;
      if (c < NN) Ob[p * NN + c] = vals[i] * inv;
    }
  }
}

extern "C" void kernel_launch(void* const* d_in, const int* in_sizes, int n_in,
                              void* d_out, int out_size, void* d_ws, size_t ws_size,
                              hipStream_t stream) {
  (void)n_in; (void)out_size; (void)ws_size;
  const float* enc   = (const float*)d_in[0];
  const float* lastn = (const float*)d_in[1];
  const float* attr  = (const float*)d_in[2];
  const float* mask  = (const float*)d_in[3];
  const float* Wq    = (const float*)d_in[4];
  const float* Wk    = (const float*)d_in[5];
  const float* Wv    = (const float*)d_in[6];
  const float* Wc    = (const float*)d_in[7];
  const float* bc    = (const float*)d_in[8];
  float* out = (float*)d_out;
  _Float16* wsW = (_Float16*)d_ws;  // needs WS_TOTAL*2 = 131072 bytes

  const int B = in_sizes[2] / POMO;  // attr is [B, POMO, 1]

  // 5 x (112x128) + 3 x (128x128) f16 + 112x116 f32 = 293,632 bytes
  const size_t shmem =
      (size_t)(PAD_R * DD * 5 + DD * DD * 3) * sizeof(_Float16) +
      (size_t)PAD_R * SSTR * sizeof(float);

  (void)hipFuncSetAttribute((const void*)pomo_decoder_wmma,
                            hipFuncAttributeMaxDynamicSharedMemorySize,
                            (int)shmem);

  convert_weights_f16<<<(DD * DD + 255) / 256, 256, 0, stream>>>(Wq, Wk, Wv, Wc,
                                                                 wsW);
  pomo_decoder_wmma<<<B, 256, shmem, stream>>>(enc, lastn, attr, mask, Wq, bc,
                                               wsW, out);
}
